// SAGE_17910013624557
// MI455X (gfx1250) — matrix operand order repaired and checked by reading, not compiled, not run on hardware
//
#include <hip/hip_runtime.h>

// ---------------------------------------------------------------------------
// GraphSAGE forward for MI455X (gfx1250, wave32, WMMA).
//  - Dense GEMMs: v_wmma_f32_16x16x32_bf16, A as bf16 row-major (b128 loads),
//    B pre-packed to fragment layout, staged in LDS (ds_load_b128).
//  - SpMM: edge-parallel relaxed f32 global atomics (L2-resident destination).
// ---------------------------------------------------------------------------

typedef __attribute__((ext_vector_type(8)))  __bf16 v8bf;
typedef __attribute__((ext_vector_type(16))) __bf16 v16bf;
typedef __attribute__((ext_vector_type(8)))  float  v8f;

#define D_IN 128
#define HDIM 64
#define EMB  32

// ---------------------------------------------------------------------------
// f32 -> bf16 bulk convert (row-major layout preserved)
// ---------------------------------------------------------------------------
__global__ void cvt_f32_bf16(const float* __restrict__ x, __bf16* __restrict__ y, long n)
{
    const long i = (long)blockIdx.x * blockDim.x + threadIdx.x;
    if (i < n) y[i] = (__bf16)x[i];
}

// ---------------------------------------------------------------------------
// Pack two KDIM x NOUT f32 weight matrices into bf16 WMMA B-fragments:
//   wpack[frag][lane][e], frag = ((w*KCH)+kc)*NT + nt, 512 bf16 per frag.
//   lane n = lane&15, K = kc*32 + (lane>>4)*8 + (e&7) + (e>>3)*16
// ---------------------------------------------------------------------------
template<int KDIM, int NOUT>
__global__ void pack_weights(const float* __restrict__ W0,
                             const float* __restrict__ W1,
                             __bf16* __restrict__ wpack)
{
    constexpr int KCH = KDIM / 32;
    constexpr int NT  = NOUT / 16;
    const int t = blockIdx.x * blockDim.x + threadIdx.x;
    if (t >= 2 * KCH * NT * 32) return;
    const int lane = t & 31;
    const int frag = t >> 5;
    const int nt = frag % NT;
    const int kc = (frag / NT) % KCH;
    const int w  = frag / (NT * KCH);
    const float* W = w ? W1 : W0;
    const int n     = nt * 16 + (lane & 15);
    const int kbase = kc * 32 + (lane >> 4) * 8;
    __bf16* out = wpack + (long)frag * 512 + lane * 16;
#pragma unroll
    for (int e = 0; e < 16; ++e)
        out[e] = (__bf16)W[(kbase + (e & 7) + (e >> 3) * 16) * NOUT + n];
}

// ---------------------------------------------------------------------------
// Fused dual-weight GEMM. 256 threads = 8 waves; each wave owns a 16-row
// M-tile; both packed weight matrices staged in LDS and shared by all waves.
//   ON = X @ W0 (+biasN),  OG = X @ W1 (+biasG)
//
// A fragment (bf16 16x32 tile, wave32): lane m = lane&15, g = lane>>4:
//   elems 0..7  -> K = kc*32 + g*8 + e        (contiguous 16B)
//   elems 8..15 -> K = kc*32 + 16 + g*8 + e   (contiguous 16B)
// C/D: row M = g*8 + r, col N = lane&15.
// ---------------------------------------------------------------------------
template<int KDIM, int NOUT, bool BIAS>
__global__ __launch_bounds__(256)
void sage_gemm(const __bf16* __restrict__ Xbf,
               const __bf16* __restrict__ wpack,
               const float* __restrict__ biasN, const float* __restrict__ biasG,
               float* __restrict__ ON, float* __restrict__ OG, int nrows)
{
    constexpr int KCH = KDIM / 32;
    constexpr int NT  = NOUT / 16;
    __shared__ alignas(32) __bf16 lds[2 * KCH * NT * 512];

    // ---- cooperative global -> LDS stage of packed weights (16B/thread) ----
    constexpr int NVEC = (2 * KCH * NT * 512 * 2) / 16;   // uint4 count
    {
        const uint4* src = (const uint4*)wpack;
        uint4*       dst = (uint4*)lds;
#pragma unroll
        for (int i = threadIdx.x; i < NVEC; i += 256) dst[i] = src[i];
    }
    __syncthreads();

    const int  wave = threadIdx.x >> 5;
    const int  lane = threadIdx.x & 31;
    const int  m    = lane & 15;
    const int  g    = lane >> 4;
    const long row0 = ((long)blockIdx.x * 8 + wave) * 16;

    long rowA = row0 + m;
    if (rowA >= nrows) rowA = nrows - 1;              // clamp ragged tail
    const __bf16* xrow = Xbf + rowA * KDIM;

    v8f acc[2][NT];
#pragma unroll
    for (int w = 0; w < 2; ++w)
#pragma unroll
        for (int t = 0; t < NT; ++t)
            acc[w][t] = (v8f){0.f, 0.f, 0.f, 0.f, 0.f, 0.f, 0.f, 0.f};

#pragma unroll
    for (int kc = 0; kc < KCH; ++kc) {
        if (kc + 1 < KCH)
            __builtin_prefetch(xrow + (kc + 1) * 32, 0, 1);

        const v8bf lo = *(const v8bf*)(xrow + kc * 32 + g * 8);
        const v8bf hi = *(const v8bf*)(xrow + kc * 32 + 16 + g * 8);
        const v16bf a = __builtin_shufflevector(lo, hi,
            0, 1, 2, 3, 4, 5, 6, 7, 8, 9, 10, 11, 12, 13, 14, 15);

#pragma unroll
        for (int w = 0; w < 2; ++w) {
#pragma unroll
            for (int t = 0; t < NT; ++t) {
                const v16bf b = *(const v16bf*)
                    &lds[(((w * KCH) + kc) * NT + t) * 512 + lane * 16];
                acc[w][t] = __builtin_amdgcn_wmma_f32_16x16x32_bf16(
                    false, a, false, b, (short)0, acc[w][t], false, false);
            }
        }
    }

    // ---- store D (+ optional bias) with tail bounds check ----
#pragma unroll
    for (int w = 0; w < 2; ++w) {
        float* O = w ? OG : ON;
        const float* bias = w ? biasG : biasN;
#pragma unroll
        for (int t = 0; t < NT; ++t) {
            const int   n    = t * 16 + m;
            const float badd = BIAS ? bias[n] : 0.0f;
#pragma unroll
            for (int r = 0; r < 8; ++r) {
                const long orow = row0 + g * 8 + r;
                if (orow < nrows)
                    O[orow * NOUT + n] = acc[w][t][r] + badd;
            }
        }
    }
}

// ---------------------------------------------------------------------------
// Zero-fill
// ---------------------------------------------------------------------------
__global__ void zero_f32(float* __restrict__ p, long n)
{
    const long i = (long)blockIdx.x * blockDim.x + threadIdx.x;
    if (i < n) p[i] = 0.0f;
}

// ---------------------------------------------------------------------------
// SpMM: agg[rows[e], :] += vals[e] * x[cols[e], :]
// ---------------------------------------------------------------------------
template<int HH>
__global__ __launch_bounds__(256)
void sage_spmm(const int* __restrict__ rows, const int* __restrict__ cols,
               const float* __restrict__ vals, const float* __restrict__ xn,
               float* __restrict__ agg, int E)
{
    const int e = blockIdx.x * blockDim.y + threadIdx.y;
    if (e >= E) return;
    const int   r = rows[e];
    const int   c = cols[e];
    const float v = vals[e];
    const int   f = threadIdx.x;
    const float contrib = v * xn[(long)c * HH + f];
    __hip_atomic_fetch_add(&agg[(long)r * HH + f], contrib,
                           __ATOMIC_RELAXED, __HIP_MEMORY_SCOPE_AGENT);
}

// ---------------------------------------------------------------------------
// z = x_node + agg + ab ; accumulate column sum / sumsq into stats.
// ---------------------------------------------------------------------------
__global__ void sage_stats(const float* __restrict__ xn,
                           const float* __restrict__ agg,
                           const float* __restrict__ ab,
                           float* __restrict__ z,
                           float* __restrict__ stats,
                           int n_rows, int H)
{
    const int   col = threadIdx.x;
    const float abv = ab[col];
    float s = 0.0f, ss = 0.0f;
    for (long r = blockIdx.x; r < n_rows; r += gridDim.x) {
        const long  o  = r * H + col;
        const float zz = xn[o] + agg[o] + abv;
        z[o] = zz;
        s  += zz;
        ss += zz * zz;
    }
    __hip_atomic_fetch_add(&stats[col], s,
                           __ATOMIC_RELAXED, __HIP_MEMORY_SCOPE_AGENT);
    __hip_atomic_fetch_add(&stats[H + col], ss,
                           __ATOMIC_RELAXED, __HIP_MEMORY_SCOPE_AGENT);
}

// ---------------------------------------------------------------------------
// h(bf16) = relu((z - mean) * rsqrt(var + eps) * gamma + beta)
// bf16 output feeds the next WMMA GEMM directly.
// ---------------------------------------------------------------------------
__global__ void sage_bn_relu_bf16(const float* __restrict__ z,
                                  const float* __restrict__ stats,
                                  const float* __restrict__ gma,
                                  const float* __restrict__ bta,
                                  __bf16* __restrict__ h,
                                  long total, int H, float invN)
{
    const long i = (long)blockIdx.x * blockDim.x + threadIdx.x;
    if (i >= total) return;
    const int   col = (int)(i & (H - 1));
    const float mn  = stats[col] * invN;
    const float var = stats[H + col] * invN - mn * mn;
    const float y   = (z[i] - mn) * rsqrtf(var + 1e-5f) * gma[col] + bta[col];
    h[i] = (__bf16)fmaxf(y, 0.0f);
}

// ---------------------------------------------------------------------------
// out[i, :] = x_node[idx[i], :] + agg[idx[i], :] + abL
// ---------------------------------------------------------------------------
__global__ void sage_gather(const float* __restrict__ xn,
                            const float* __restrict__ agg,
                            const float* __restrict__ ab,
                            const int* __restrict__ idx,
                            float* __restrict__ out, int nidx)
{
    const int i = blockIdx.x * blockDim.x + threadIdx.x;
    if (i >= nidx * EMB) return;
    const int  row = i >> 5;              // EMB == 32
    const int  col = i & (EMB - 1);
    const long src = (long)idx[row] * EMB + col;
    out[i] = xn[src] + agg[src] + ab[col];
}

// ---------------------------------------------------------------------------
// Host-side orchestration
// ---------------------------------------------------------------------------
extern "C" void kernel_launch(void* const* d_in, const int* in_sizes, int n_in,
                              void* d_out, int out_size, void* d_ws, size_t ws_size,
                              hipStream_t stream)
{
    const float* features = (const float*)d_in[0];
    const int*   rows     = (const int*)  d_in[1];
    const int*   cols     = (const int*)  d_in[2];
    const float* vals     = (const float*)d_in[3];
    const int*   idx      = (const int*)  d_in[4];
    const float* Wn0 = (const float*)d_in[5];
    const float* bn0 = (const float*)d_in[6];
    const float* Wg0 = (const float*)d_in[7];
    const float* bg0 = (const float*)d_in[8];
    const float* ab0 = (const float*)d_in[9];
    const float* g0  = (const float*)d_in[10];
    const float* b0  = (const float*)d_in[11];
    const float* Wn1 = (const float*)d_in[12];
    const float* Wg1 = (const float*)d_in[13];
    const float* ab1 = (const float*)d_in[14];
    const float* g1  = (const float*)d_in[15];
    const float* b1  = (const float*)d_in[16];
    const float* WnL = (const float*)d_in[17];
    const float* WgL = (const float*)d_in[18];
    const float* abL = (const float*)d_in[19];

    const int N    = in_sizes[0] / D_IN;   // 100000 (multiple of 16)
    const int E    = in_sizes[1];          // 800000
    const int NIDX = in_sizes[4];          // 50000
    const float invN = 1.0f / (float)N;

    // ---- workspace layout ----
    float*  A     = (float*)d_ws;                 // N x 64 f32
    float*  B     = A + (long)N * HDIM;           // N x 64 f32
    float*  C     = B + (long)N * HDIM;           // N x 64 f32
    float*  stats = C + (long)N * HDIM;           // 128 f32
    __bf16* Xbf   = (__bf16*)(stats + 128);       // N x 128 bf16 (reused as h)
    __bf16* wpack = Xbf + (long)N * D_IN;         // packed weight fragments

    const long NH    = (long)N * HDIM;
    const int  zgrid = (int)((NH + 255) / 256);
    const int  sgrid = 512;
    const unsigned ggrid = (unsigned)((N / 16 + 7) / 8);   // 8 waves/block

    // ===================== Layer 0 =====================
    cvt_f32_bf16<<<(int)(((long)N * D_IN + 255) / 256), 256, 0, stream>>>(
        features, Xbf, (long)N * D_IN);
    pack_weights<D_IN, HDIM><<<4, 256, 0, stream>>>(Wn0, Wg0, wpack);
    sage_gemm<D_IN, HDIM, true><<<ggrid, 256, 0, stream>>>(
        Xbf, wpack, bn0, bg0, A, B, N);                 // node->A, neigh->B
    zero_f32<<<zgrid, 256, 0, stream>>>(C, NH);
    zero_f32<<<1, 128, 0, stream>>>(stats, 2 * HDIM);
    sage_spmm<HDIM><<<(E + 3) / 4, dim3(HDIM, 4), 0, stream>>>(
        rows, cols, vals, B, C, E);                     // agg -> C
    sage_stats<<<sgrid, HDIM, 0, stream>>>(A, C, ab0, B, stats, N, HDIM);  // z -> B
    sage_bn_relu_bf16<<<zgrid, 256, 0, stream>>>(B, stats, g0, b0,
                                                 Xbf, NH, HDIM, invN);     // h0 -> Xbf

    // ===================== Layer 1 =====================
    pack_weights<HDIM, HDIM><<<2, 256, 0, stream>>>(Wn1, Wg1, wpack);
    sage_gemm<HDIM, HDIM, false><<<ggrid, 256, 0, stream>>>(
        Xbf, wpack, nullptr, nullptr, B, C, N);         // node->B, neigh->C
    zero_f32<<<zgrid, 256, 0, stream>>>(A, NH);
    zero_f32<<<1, 128, 0, stream>>>(stats, 2 * HDIM);
    sage_spmm<HDIM><<<(E + 3) / 4, dim3(HDIM, 4), 0, stream>>>(
        rows, cols, vals, C, A, E);                     // agg -> A
    sage_stats<<<sgrid, HDIM, 0, stream>>>(B, A, ab1, B, stats, N, HDIM);  // z -> B
    sage_bn_relu_bf16<<<zgrid, 256, 0, stream>>>(B, stats, g1, b1,
                                                 Xbf, NH, HDIM, invN);     // h1 -> Xbf

    // ===================== conv_last =====================
    pack_weights<HDIM, EMB><<<1, 256, 0, stream>>>(WnL, WgL, wpack);
    sage_gemm<HDIM, EMB, false><<<ggrid, 256, 0, stream>>>(
        Xbf, wpack, nullptr, nullptr, B, C, N);         // node->B, neigh->C
    const long NE = (long)N * EMB;
    zero_f32<<<(int)((NE + 255) / 256), 256, 0, stream>>>(A, NE);
    sage_spmm<EMB><<<(E + 7) / 8, dim3(EMB, 8), 0, stream>>>(
        rows, cols, vals, C, A, E);
    sage_gather<<<(NIDX * EMB + 255) / 256, 256, 0, stream>>>(
        B, A, abL, idx, (float*)d_out, NIDX);
}
